// tamlstm_68358699483157
// MI455X (gfx1250) — compile-verified, matrix operand order checked by reading, not run on hardware
//
#include <hip/hip_runtime.h>

// Problem dims (fixed by the reference)
#define B_   256
#define T_   512
#define D_   64
#define L_   32
#define H_   128
#define G4H  512          // 4*H
#define KTOT 192          // D + H (fused [x_t | h] K dimension)

typedef _Float16 h16;
typedef __attribute__((ext_vector_type(16))) _Float16 v16h;
typedef __attribute__((ext_vector_type(8)))  _Float16 v8h;
typedef __attribute__((ext_vector_type(8)))  float    v8f;

// Raw workgroup split-barrier WITHOUT the full acquire-release fence of
// __syncthreads(): only DScnt must reach 0 for LDS-mediated communication.
// LOADcnt/STOREcnt stay outstanding so global prefetches / out-stores span it.
__device__ __forceinline__ void wg_barrier_lds_only() {
  asm volatile("s_wait_dscnt 0x0\n\t"
               "s_barrier_signal -1\n\t"
               "s_barrier_wait -1" ::: "memory");
}

// Load 8 consecutive f32 and convert to a v8h half-chunk.
__device__ __forceinline__ v8h cvt8(const float* __restrict__ p) {
  float4 a = *(const float4*)p;
  float4 b = *(const float4*)(p + 4);
  v8h r;
  r[0] = (h16)a.x; r[1] = (h16)a.y; r[2] = (h16)a.z; r[3] = (h16)a.w;
  r[4] = (h16)b.x; r[5] = (h16)b.y; r[6] = (h16)b.z; r[7] = (h16)b.w;
  return r;
}

__device__ __forceinline__ float sigm(float x) {
  return 1.f / (1.f + __expf(-x));
}
__device__ __forceinline__ float tanh_fast(float x) {
  const float e = __expf(-2.f * fabsf(x));
  return copysignf((1.f - e) / (1.f + e), x);
}

// ---------------------------------------------------------------------------
// Prep: static gate bias z[b,n] = b[n] + x_static[b,:] . W_zh[n,:]  (K=32)
// ---------------------------------------------------------------------------
__global__ void prep_static(const float* __restrict__ x_static,
                            const float* __restrict__ W_zh,
                            const float* __restrict__ bias,
                            float* __restrict__ z) {
  int i = blockIdx.x * blockDim.x + threadIdx.x;      // over 256*512
  if (i >= B_ * G4H) return;
  int b = i / G4H, n = i % G4H;
  float acc = bias[n];
#pragma unroll
  for (int l = 0; l < L_; ++l)
    acc += x_static[b * L_ + l] * W_zh[n * L_ + l];
  z[i] = acc;
}

// ---------------------------------------------------------------------------
// Recurrent kernel. Grid = 16 WGs (16 batch rows each), block = 256 (8 waves).
// Wave w owns gate N-tiles {w, w+8, w+16, w+24}: its 4 accumulators hold the
// matching (i,f,g,o) values of hidden columns 16w..16w+15, so the LSTM
// nonlinearity runs ENTIRELY on WMMA accumulator registers — no gates buffer.
// Weights (f16) are register-resident for all 512 steps (4x6 v16h per wave).
// A panel + f32 h copy are double-buffered in LDS -> ONE barrier per step:
//   phase 1: stage x_{t+1} into buf[nxt], prefetch x_{t+2}
//   phase 2: hoist all 6 A-tiles (12x ds_load_b128, staged dscnt waits), then
//            6 k-steps x 4 chains of v_wmma_f32_16x16x32_f16;
//            lazy head out[:,t-1] from hbuf[cur] co-executes with the WMMAs
//   phase 3: gates -> c (registers) -> h; h to buf[nxt] (f16) + hbuf[nxt]
//   barrier (dscnt-only)
// Epilogue emits out[:,T-1].
// ---------------------------------------------------------------------------
__global__ __launch_bounds__(256, 1)
void lstm_scan(const float* __restrict__ x_dyn,   // [B,T,D] f32
               const float* __restrict__ W_ih,    // [4H,D]  f32
               const float* __restrict__ W_hh,    // [4H,H]  f32
               const float* __restrict__ z,       // [B,4H]  f32
               const float* __restrict__ W_out,   // [1,H]
               const float* __restrict__ b_out,   // [1]
               float* __restrict__ out) {         // [B,T]
  __shared__ h16   Abuf[2][16 * KTOT];            // 2 x 6144 B (x_t | h panel)
  __shared__ float hbuf[2][16 * H_];              // 2 x 8192 B (f32 h for head)

  const int tid  = threadIdx.x;
  const int wave = tid >> 5;          // 0..7
  const int lane = tid & 31;          // wave32
  const int bg   = blockIdx.x * 16;   // batch base for this WGP

  // A/B 16-bit WMMA lane pattern: lane L -> row (L&15), k-chunk base 8*(L>=16)
  const int lrow  = lane & 15;
  const int kbase = (lane >> 4) * 8;
  const int mhalf = kbase;            // C/D layout: vgpr r -> row mhalf + r
  const int j     = wave * 16 + lrow; // hidden column this lane owns in phase 3

  // ---- one-time: register-resident B cache (f32 -> f16), 4 tiles x 6 ks ----
  // tile nt covers gate columns nt*128 + 16w .. +15  (i/f/g/o for nt=0..3)
  v16h bc[4][6];
#pragma unroll
  for (int nt = 0; nt < 4; ++nt) {
    const int n = nt * H_ + j;                     // W row (gate index)
#pragma unroll
    for (int ks = 0; ks < 6; ++ks) {
#pragma unroll
      for (int c = 0; c < 2; ++c) {
        const int kg = ks * 32 + kbase + c * 16;
        const float* s = (kg < D_) ? (W_ih + n * D_ + kg)
                                   : (W_hh + n * H_ + (kg - D_));
        ((v8h*)&bc[nt][ks])[c] = cvt8(s);
      }
    }
  }

  // static gate bias tiles (C/D layout)
  v8f zr[4];
#pragma unroll
  for (int nt = 0; nt < 4; ++nt)
#pragma unroll
    for (int r = 0; r < 8; ++r)
      zr[nt][r] = z[(size_t)(bg + mhalf + r) * G4H + nt * H_ + j];

  // output head slice + cell state (8 rows x 1 col per lane) in registers
  const float wreg0 = W_out[lane],      wreg1 = W_out[lane + 32],
              wreg2 = W_out[lane + 64], wreg3 = W_out[lane + 96];
  const float bo = b_out[0];
  v8f cst;
#pragma unroll
  for (int r = 0; r < 8; ++r) cst[r] = 0.f;

  // zero h-region of Abuf[0] (h_{-1} = 0)
  for (int i = tid; i < 16 * H_; i += 256)
    Abuf[0][(i >> 7) * KTOT + D_ + (i & 127)] = (h16)0.f;

  // x staging: thread owns 4 consecutive cols of one row (256 thr == 16x16)
  const int xrow = tid >> 4, xcol = (tid & 15) * 4;
  const float4* xp = (const float4*)(x_dyn + (size_t)(bg + xrow) * T_ * D_ + xcol);
  {
    float4 x0 = xp[0];                              // stage x_0 into buf 0
    union { uint2 u; h16 h[4]; } pk;
    pk.h[0] = (h16)x0.x; pk.h[1] = (h16)x0.y;
    pk.h[2] = (h16)x0.z; pk.h[3] = (h16)x0.w;
    *(uint2*)(&Abuf[0][xrow * KTOT + xcol]) = pk.u;
  }
  float4 xreg = xp[D_ / 4];                         // prefetch x_1

  __syncthreads();   // one-time full barrier (init)

  for (int t = 0; t < T_; ++t) {
    const int cur = t & 1, nxt = cur ^ 1;

    // -- phase 1: stage x_{t+1} into buf[nxt]; prefetch x_{t+2} --
    {
      union { uint2 u; h16 h[4]; } pk;
      pk.h[0] = (h16)xreg.x; pk.h[1] = (h16)xreg.y;
      pk.h[2] = (h16)xreg.z; pk.h[3] = (h16)xreg.w;
      *(uint2*)(&Abuf[nxt][xrow * KTOT + xcol]) = pk.u;
      const int tn = (t + 2 < T_) ? (t + 2) : (T_ - 1);  // clamped dummy
      xreg = xp[(size_t)tn * (D_ / 4)];
    }

    // -- phase 2: hoist all A-tiles, then 4 accumulator chains x 6 k-steps --
    v16h av[6];
#pragma unroll
    for (int ks = 0; ks < 6; ++ks) {
      const int kb = ks * 32 + kbase;
      ((v8h*)&av[ks])[0] = *(const v8h*)(&Abuf[cur][lrow * KTOT + kb]);
      ((v8h*)&av[ks])[1] = *(const v8h*)(&Abuf[cur][lrow * KTOT + kb + 16]);
    }
    v8f acc0 = zr[0], acc1 = zr[1], acc2 = zr[2], acc3 = zr[3];
#pragma unroll
    for (int ks = 0; ks < 6; ++ks) {
      acc0 = __builtin_amdgcn_wmma_f32_16x16x32_f16(false, av[ks], false, bc[0][ks],
                                                    (short)0, acc0, false, false);
      acc1 = __builtin_amdgcn_wmma_f32_16x16x32_f16(false, av[ks], false, bc[1][ks],
                                                    (short)0, acc1, false, false);
      acc2 = __builtin_amdgcn_wmma_f32_16x16x32_f16(false, av[ks], false, bc[2][ks],
                                                    (short)0, acc2, false, false);
      acc3 = __builtin_amdgcn_wmma_f32_16x16x32_f16(false, av[ks], false, bc[3][ks],
                                                    (short)0, acc3, false, false);
    }

    // -- lazy head: out[:, t-1] from hbuf[cur]; co-executes with WMMA drain --
    if (t > 0) {
      const float* hb = &hbuf[cur][0];
      const int r0 = wave * 2, r1 = r0 + 1;
      float pa = hb[r0 * H_ + lane]      * wreg0 + hb[r0 * H_ + lane + 32] * wreg1
               + hb[r0 * H_ + lane + 64] * wreg2 + hb[r0 * H_ + lane + 96] * wreg3;
      float pb = hb[r1 * H_ + lane]      * wreg0 + hb[r1 * H_ + lane + 32] * wreg1
               + hb[r1 * H_ + lane + 64] * wreg2 + hb[r1 * H_ + lane + 96] * wreg3;
      pa += __shfl_down(pa, 16); pb += __shfl_down(pb, 16);
      pa += __shfl_down(pa, 8);  pb += __shfl_down(pb, 8);
      pa += __shfl_down(pa, 4);  pb += __shfl_down(pb, 4);
      pa += __shfl_down(pa, 2);  pb += __shfl_down(pb, 2);
      pa += __shfl_down(pa, 1);  pb += __shfl_down(pb, 1);
      if (lane == 0) {
        out[(size_t)(bg + r0) * T_ + (t - 1)] = pa + bo;
        out[(size_t)(bg + r1) * T_ + (t - 1)] = pb + bo;
      }
    }

    // -- phase 3: LSTM nonlinearity on accumulator registers --
    // acc0/1/2/3[r] are (i,f,g,o) of element (row mhalf+r, hidden col j).
#pragma unroll
    for (int r = 0; r < 8; ++r) {
      const float si = sigm(acc0[r]);
      const float sf = sigm(acc1[r]);
      const float tg = tanh_fast(acc2[r]);
      const float so = sigm(acc3[r]);
      const float c  = sf * cst[r] + si * tg;
      const float h  = so * tanh_fast(c);
      cst[r] = c;
      Abuf[nxt][(mhalf + r) * KTOT + D_ + j] = (h16)h;  // next step's A panel
      hbuf[nxt][(mhalf + r) * H_ + j] = h;              // f32 copy for the head
    }

    wg_barrier_lds_only();   // single per-step barrier (dscnt-only)
  }

  // -- epilogue head: out[:, T-1] from hbuf[T_ & 1] --
  {
    const float* hb = &hbuf[T_ & 1][0];
    const int r0 = wave * 2, r1 = r0 + 1;
    float pa = hb[r0 * H_ + lane]      * wreg0 + hb[r0 * H_ + lane + 32] * wreg1
             + hb[r0 * H_ + lane + 64] * wreg2 + hb[r0 * H_ + lane + 96] * wreg3;
    float pb = hb[r1 * H_ + lane]      * wreg0 + hb[r1 * H_ + lane + 32] * wreg1
             + hb[r1 * H_ + lane + 64] * wreg2 + hb[r1 * H_ + lane + 96] * wreg3;
    pa += __shfl_down(pa, 16); pb += __shfl_down(pb, 16);
    pa += __shfl_down(pa, 8);  pb += __shfl_down(pb, 8);
    pa += __shfl_down(pa, 4);  pb += __shfl_down(pb, 4);
    pa += __shfl_down(pa, 2);  pb += __shfl_down(pb, 2);
    pa += __shfl_down(pa, 1);  pb += __shfl_down(pb, 1);
    if (lane == 0) {
      out[(size_t)(bg + r0) * T_ + (T_ - 1)] = pa + bo;
      out[(size_t)(bg + r1) * T_ + (T_ - 1)] = pb + bo;
    }
  }
}

// ---------------------------------------------------------------------------
extern "C" void kernel_launch(void* const* d_in, const int* in_sizes, int n_in,
                              void* d_out, int out_size, void* d_ws, size_t ws_size,
                              hipStream_t stream) {
  (void)in_sizes; (void)n_in; (void)out_size; (void)ws_size;
  const float* x_dyn = (const float*)d_in[0];
  const float* x_sta = (const float*)d_in[1];
  const float* W_ih  = (const float*)d_in[2];
  const float* W_hh  = (const float*)d_in[3];
  const float* W_zh  = (const float*)d_in[4];
  const float* bias  = (const float*)d_in[5];
  const float* W_out = (const float*)d_in[6];
  const float* b_out = (const float*)d_in[7];
  float* out = (float*)d_out;

  // workspace: z = static gate bias, [B,4H] f32 (512 KB)
  float* z = (float*)d_ws;

  prep_static<<<(B_ * G4H + 255) / 256, 256, 0, stream>>>(x_sta, W_zh, bias, z);
  lstm_scan<<<B_ / 16, 256, 0, stream>>>(x_dyn, W_ih, W_hh, z, W_out, b_out, out);
}
